// newMM_77180562309399
// MI455X (gfx1250) — matrix operand order repaired and checked by reading, not compiled, not run on hardware
//
#include <hip/hip_runtime.h>
#include <hip/hip_bf16.h>

typedef __attribute__((ext_vector_type(16))) _Float16 v16h;
typedef __attribute__((ext_vector_type(8)))  _Float16 v8h;
typedef __attribute__((ext_vector_type(8)))  float    v8f;

// ---------------------------------------------------------------------------
// Kernel 1: forward of the straight-through estimator is exactly sign(h).
// Build A (rowsPad x 64) fp16: row r = b*5 + g, col cc in [0,45) maps to
// flat j = g*45+cc of the 15x15 matrix (c = j/15, k = j%15). cols 45..63 = 0.
// ---------------------------------------------------------------------------
__global__ void sign_pack_kernel(const float* __restrict__ x,
                                 const float* __restrict__ S,
                                 const float* __restrict__ T,
                                 _Float16* __restrict__ A,
                                 int rows, int totalElems) {
    int gid = blockIdx.x * blockDim.x + threadIdx.x;
    if (gid >= totalElems) return;
    int col = gid & 63;
    int row = gid >> 6;
    _Float16 v = (_Float16)0.0f;
    if (col < 45 && row < rows) {
        int b = row / 5;
        int g = row - b * 5;
        int j = g * 45 + col;          // 0..224 in the flattened 15x15
        int c = j / 15;
        int k = j - c * 15;
        float h = x[b * 30 + 2 * c]     * S[c * 30 + k]
                + x[b * 30 + 2 * c + 1] * S[c * 30 + 15 + k]
                - T[c * 15 + k] - 1e-4f;
        v = (h > 0.0f) ? (_Float16)1.0f : ((h < 0.0f) ? (_Float16)(-1.0f) : (_Float16)0.0f);
    }
    A[(size_t)row * 64 + col] = v;
}

// ---------------------------------------------------------------------------
// Kernel 2: pre-swizzle H (45x4096 fp32) into fp16 B-fragment layout, K->64.
// Hp[((nt*2+ks)*32 + lane)*16 + h] = H[K][N],
//   K = ks*32 + 16*(h>>3) + 8*(lane>>4) + (h&7),  N = nt*16 + (lane&15)   (ISA 7.12.2 B layout)
// ---------------------------------------------------------------------------
__global__ void pack_H_kernel(const float* __restrict__ H,
                              _Float16* __restrict__ Hp) {
    int gid = blockIdx.x * blockDim.x + threadIdx.x;   // 0 .. 256*2*32*16-1
    int h    = gid & 15;
    int lane = (gid >> 4) & 31;
    int t    = gid >> 9;          // nt*2 + ks
    int ks   = t & 1;
    int nt   = t >> 1;
    int K = ks * 32 + ((h >> 3) << 4) + ((lane >> 4) << 3) + (h & 7);
    int N = nt * 16 + (lane & 15);
    float v = (K < 45) ? H[K * 4096 + N] : 0.0f;
    Hp[gid] = (_Float16)v;
}

// ---------------------------------------------------------------------------
// Kernel 3: one wave handles FOUR 16-row M-tiles (A is only 16 VGPRs/tile) so
// each 64B/lane B-fragment load feeds 8 WMMAs -> 4x less L2 traffic. B loads
// are software-pipelined (preload nt+1 while computing nt; last iter peeled).
// Fused running argmax over N=4096, butterfly lane reduce, LUT gather.
// ---------------------------------------------------------------------------
__global__ void __launch_bounds__(256)
gemm_argmax_kernel(const _Float16* __restrict__ A,
                   const _Float16* __restrict__ Hp,
                   const float* __restrict__ LUT,
                   float* __restrict__ out,
                   int rows, int mtiles) {
    const int wave  = threadIdx.x >> 5;
    const int lane  = threadIdx.x & 31;
    const int group = blockIdx.x * 8 + wave;   // group of 4 M-tiles
    const int tbase = group * 4;
    if (tbase >= mtiles) return;

    const int m  = lane & 15;   // row within tile (A), N residue (B/C)
    const int hi = lane >> 4;

    // --- A fragments per ISA "16-bit A-Matrix 16x32":
    // halves 0..7  <-> K = 8*hi + 0..7 ; halves 8..15 <-> K = 16 + 8*hi + 0..7
    v16h a0[4], a1[4];
#pragma unroll
    for (int t = 0; t < 4; ++t) {
        int mt = tbase + t;
        if (mt >= mtiles) mt = mtiles - 1;     // clamp (stores guarded later)
        const _Float16* arow = A + ((size_t)mt * 16 + m) * 64;
        v8h p0 = *(const v8h*)(arow + 8 * hi);
        v8h p1 = *(const v8h*)(arow + 16 + 8 * hi);
        v8h p2 = *(const v8h*)(arow + 32 + 8 * hi);
        v8h p3 = *(const v8h*)(arow + 48 + 8 * hi);
#pragma unroll
        for (int i = 0; i < 8; ++i) {
            a0[t][i] = p0[i]; a0[t][i + 8] = p1[i];
            a1[t][i] = p2[i]; a1[t][i + 8] = p3[i];
        }
    }

    float maxv[4][8];
    int   maxi[4][8];
#pragma unroll
    for (int t = 0; t < 4; ++t)
#pragma unroll
        for (int j = 0; j < 8; ++j) { maxv[t][j] = -__builtin_inff(); maxi[t][j] = 0; }

    // B fragment base for this lane; stride per N-tile = 2*32*16 = 1024 halves
    const _Float16* bptr = Hp + (size_t)lane * 16;
    v16h b0 = *(const v16h*)(bptr);
    v16h b1 = *(const v16h*)(bptr + 512);

    for (int nt = 0; nt < 255; ++nt) {
        const _Float16* nxt = bptr + (size_t)(nt + 1) * 1024;
        v16h nb0 = *(const v16h*)(nxt);
        v16h nb1 = *(const v16h*)(nxt + 512);
        const int n = nt * 16 + m;
#pragma unroll
        for (int t = 0; t < 4; ++t) {
            v8f c = {};
            c = __builtin_amdgcn_wmma_f32_16x16x32_f16(false, a0[t], false, b0,
                                                       (short)0, c, false, false);
            c = __builtin_amdgcn_wmma_f32_16x16x32_f16(false, a1[t], false, b1,
                                                       (short)0, c, false, false);
#pragma unroll
            for (int j = 0; j < 8; ++j) {
                // strictly-greater keeps the first (lowest-n) max within a lane
                if (c[j] > maxv[t][j]) { maxv[t][j] = c[j]; maxi[t][j] = n; }
            }
        }
        b0 = nb0; b1 = nb1;
    }
    {   // peeled last N-tile (nt = 255)
        const int n = 255 * 16 + m;
#pragma unroll
        for (int t = 0; t < 4; ++t) {
            v8f c = {};
            c = __builtin_amdgcn_wmma_f32_16x16x32_f16(false, a0[t], false, b0,
                                                       (short)0, c, false, false);
            c = __builtin_amdgcn_wmma_f32_16x16x32_f16(false, a1[t], false, b1,
                                                       (short)0, c, false, false);
#pragma unroll
            for (int j = 0; j < 8; ++j) {
                if (c[j] > maxv[t][j]) { maxv[t][j] = c[j]; maxi[t][j] = n; }
            }
        }
    }

    // --- reduce across the 16 lanes of each C-matrix half; rows = 8*hi + j.
#pragma unroll
    for (int t = 0; t < 4; ++t) {
        const int mt = tbase + t;
#pragma unroll
        for (int j = 0; j < 8; ++j) {
            float v   = maxv[t][j];
            int   idx = maxi[t][j];
#pragma unroll
            for (int off = 1; off < 16; off <<= 1) {
                float v2 = __shfl_xor(v, off, 32);
                int   i2 = __shfl_xor(idx, off, 32);
                if (v2 > v || (v2 == v && i2 < idx)) { v = v2; idx = i2; }
            }
            // every lane in the half now holds the tile-row argmax
            int r = mt * 16 + hi * 8 + j;      // global row in (B*5)
            if (mt < mtiles && r < rows) {
                int g = r % 5;
                const float* src = LUT + ((size_t)g * 4096 + idx) * 32;
                float*       dst = out + (size_t)r * 32;
                float2 val = *(const float2*)(src + m * 2);  // 16 lanes x 8B = 128B
                *(float2*)(dst + m * 2) = val;
            }
        }
    }
}

extern "C" void kernel_launch(void* const* d_in, const int* in_sizes, int n_in,
                              void* d_out, int out_size, void* d_ws, size_t ws_size,
                              hipStream_t stream) {
    const float* x   = (const float*)d_in[0];   // (B, 30)
    const float* S   = (const float*)d_in[1];   // (15, 2, 15)
    const float* H   = (const float*)d_in[2];   // (45, 4096)
    const float* T   = (const float*)d_in[3];   // (15, 15)
    const float* LUT = (const float*)d_in[4];   // (5, 4096, 32)
    float* out = (float*)d_out;

    const int B      = in_sizes[0] / 30;        // 16384
    const int rows   = B * 5;                   // 81920
    const int mtiles = (rows + 15) / 16;        // 5120

    // Workspace: A (mtiles*16 x 64 fp16) then Hpack (256*2*32*16 fp16, 512 KB)
    _Float16* Aws = (_Float16*)d_ws;
    size_t aBytes = (size_t)mtiles * 16 * 64 * sizeof(_Float16);   // ~10.5 MB
    _Float16* Hp  = (_Float16*)((char*)d_ws + aBytes);
    (void)ws_size; (void)n_in; (void)out_size;

    {   // signs -> fp16 A matrix (zero-padded K 45->64, rows -> mtiles*16)
        int total  = mtiles * 16 * 64;
        int blocks = (total + 255) / 256;
        sign_pack_kernel<<<blocks, 256, 0, stream>>>(x, S, T, Aws, rows, total);
    }
    {   // H -> fragment-swizzled fp16 (256 N-tiles * 2 k-steps)
        int total  = 256 * 2 * 32 * 16;         // 262144
        pack_H_kernel<<<total / 256, 256, 0, stream>>>(H, Hp);
    }
    {   // fused WMMA GEMM + argmax + LUT gather: 4 M-tiles/wave, 8 waves/block
        int groups = (mtiles + 3) / 4;          // 1280
        int blocks = (groups + 7) / 8;          // 160
        gemm_argmax_kernel<<<blocks, 256, 0, stream>>>(Aws, Hp, LUT, out, rows, mtiles);
    }
}
